// TFXLNetRelativeAttention_55336358641792
// MI455X (gfx1250) — compile-verified, matrix operand order checked by reading, not run on hardware
//
#include <hip/hip_runtime.h>
#include <hip/hip_bf16.h>

// ---------------------------------------------------------------------------
// XLNet relative attention on gfx1250 (MI455X).
// bf16 WMMA pipeline + async global->LDS staging (ASYNCcnt) for transposed
// operands. Q=1024, B=4, D=1024, N=16, DH=64, RLEN=2048.
// ---------------------------------------------------------------------------

#define QLEN   1024
#define BATCH  4
#define DMODEL 1024
#define NHEAD  16
#define DHEAD  64
#define SCALE_F 0.125f      // 1/sqrt(64)
#define LN_EPS 1e-12f

typedef __attribute__((ext_vector_type(16))) __bf16 v16bf;
typedef __attribute__((ext_vector_type(8)))  float  v8f;

// K index base for fragment element-pair v (0..7), wave32 16-bit A/B layout
// (CDNA5 ISA §7.12.2): lanes 0-15 hold K {0..7,16..23},
// lanes 16-31 hold K {8..15,24..31}.
__device__ __forceinline__ int kbase(int v) {
  return (v < 4) ? (2 * v) : (16 + 2 * (v - 4));
}

__device__ __forceinline__ v8f wmma_bf16(v16bf a, v16bf b, v8f c) {
  return __builtin_amdgcn_wmma_f32_16x16x32_bf16(
      /*neg_a=*/false, a, /*neg_b=*/false, b,
      /*c_mod=*/(short)0, c, /*reuse_a=*/false, /*reuse_b=*/false);
}

// ---- CDNA5 async global->LDS copy (CDNA5 ISA) -------
// One lane moves 16 bytes; tracked with ASYNCcnt. Inline asm (portable across
// ROCm 7.2 / amdgpu-toolchain builtin arity differences).
__device__ __forceinline__ void async_load_b128(unsigned lds_byte_off,
                                                const float* g) {
  asm volatile("global_load_async_to_lds_b128 %0, %1, off"
               :: "v"(lds_byte_off),
                  "v"((unsigned long long)(uintptr_t)g)
               : "memory");
}
__device__ __forceinline__ void wait_async0() {
  asm volatile("s_wait_asynccnt 0x0" ::: "memory");
}
__device__ __forceinline__ void wait_ds0() {
  asm volatile("s_wait_dscnt 0x0" ::: "memory");
}
__device__ __forceinline__ unsigned lds_off_of(const void* p) {
  // Generic (flat) pointers to LDS carry the LDS byte offset in the low 32
  // bits (ISA 10.2 aperture mapping).
  return (unsigned)(uintptr_t)p;
}

// Fragment load from a contiguous fp32 row: per-lane K elements are
// {koff+0..7, koff+16..23} -> four aligned float4 loads, convert to bf16.
// Caller passes base already offset by khalf (0 or 32). bias may be null.
__device__ __forceinline__ v16bf frag_f4(const float* base, const float* bias,
                                         int lane) {
  const int koff = (lane & 16) ? 8 : 0;
  const float4* p = reinterpret_cast<const float4*>(base + koff);
  float4 q0 = p[0];   // koff + 0..3
  float4 q1 = p[1];   // koff + 4..7
  float4 q2 = p[4];   // koff + 16..19
  float4 q3 = p[5];   // koff + 20..23
  if (bias) {
    const float4* bp = reinterpret_cast<const float4*>(bias + koff);
    float4 b0 = bp[0], b1 = bp[1], b2 = bp[4], b3 = bp[5];
    q0.x += b0.x; q0.y += b0.y; q0.z += b0.z; q0.w += b0.w;
    q1.x += b1.x; q1.y += b1.y; q1.z += b1.z; q1.w += b1.w;
    q2.x += b2.x; q2.y += b2.y; q2.z += b2.z; q2.w += b2.w;
    q3.x += b3.x; q3.y += b3.y; q3.z += b3.z; q3.w += b3.w;
  }
  v16bf f;
  f[0]  = (__bf16)q0.x; f[1]  = (__bf16)q0.y;
  f[2]  = (__bf16)q0.z; f[3]  = (__bf16)q0.w;
  f[4]  = (__bf16)q1.x; f[5]  = (__bf16)q1.y;
  f[6]  = (__bf16)q1.z; f[7]  = (__bf16)q1.w;
  f[8]  = (__bf16)q2.x; f[9]  = (__bf16)q2.y;
  f[10] = (__bf16)q2.z; f[11] = (__bf16)q2.w;
  f[12] = (__bf16)q3.x; f[13] = (__bf16)q3.y;
  f[14] = (__bf16)q3.z; f[15] = (__bf16)q3.w;
  return f;
}

// ---------------------------------------------------------------------------
// GEMM: C[m][n] = sum_k A[m][k] * B[k*ldk + n*ldn]  (+ residual)
// 8 waves/block -> 128x64 C tile. B tile (32k x 64n) is async-staged into
// LDS each k-step and shared by all waves.
// NCONTIG: true  -> B rows contiguous along n (projection weights, ldn==1)
//          false -> B contiguous along k (o_w^T, ldk==1)
// ---------------------------------------------------------------------------
template <bool NCONTIG>
__global__ __launch_bounds__(256)
void gemm_bf16_kernel(const float* __restrict__ A, long lda,
                      const float* __restrict__ Bm, long ldk, long ldn,
                      float* __restrict__ C, long ldc,
                      int M, int N, int K,
                      const float* __restrict__ residual) {
  __shared__ float bt[32 * 64];  // [k][n] if NCONTIG else [n][k]
  const int tid  = threadIdx.x;
  const int lane = tid & 31;
  const int wv   = tid >> 5;
  const int m0   = blockIdx.x * 128 + wv * 16;
  const int n0   = blockIdx.y * 64;
  if (m0 >= M || n0 >= N) return;

  v8f acc[4];
#pragma unroll
  for (int t = 0; t < 4; ++t) {
#pragma unroll
    for (int e = 0; e < 8; ++e) acc[t][e] = 0.f;
  }

  const int ncl  = lane & 15;
  const int koff = (lane & 16) ? 8 : 0;
  const int mrow = m0 + ncl;
  const unsigned btb = lds_off_of(&bt[0]);

  for (int k0 = 0; k0 < K; k0 += 32) {
    // ---- async-stage the 32x64 B tile (8KB): 8 fp32 per thread ----
    {
      const int idx = tid * 8;  // flat float index into the tile
      const float* g;
      if (NCONTIG) g = Bm + (long)(k0 + (idx >> 6)) * ldk + (n0 + (idx & 63));
      else         g = Bm + (long)(n0 + (idx >> 5)) * ldn + (k0 + (idx & 31));
      const unsigned l = btb + (unsigned)idx * 4u;
      async_load_b128(l, g);
      async_load_b128(l + 16u, g + 4);
    }
    wait_async0();
    __syncthreads();

    // ---- A fragment straight from global (contiguous K) ----
    v16bf af = frag_f4(A + (long)mrow * lda + k0, nullptr, lane);

#pragma unroll
    for (int t = 0; t < 4; ++t) {
      const int nc = t * 16 + ncl;
      v16bf bf;
#pragma unroll
      for (int v = 0; v < 8; ++v) {
        const int k = kbase(v) + koff;
        if (NCONTIG) {
          bf[2 * v]     = (__bf16)bt[k * 64 + nc];
          bf[2 * v + 1] = (__bf16)bt[(k + 1) * 64 + nc];
        } else {
          bf[2 * v]     = (__bf16)bt[nc * 32 + k];
          bf[2 * v + 1] = (__bf16)bt[nc * 32 + k + 1];
        }
      }
      acc[t] = wmma_bf16(af, bf, acc[t]);
    }
    __syncthreads();  // protect bt before next stage overwrites it
  }

  const int hi8 = (lane & 16) ? 8 : 0;
#pragma unroll
  for (int t = 0; t < 4; ++t) {
#pragma unroll
    for (int e = 0; e < 8; ++e) {
      long row = m0 + e + hi8;
      long col = n0 + t * 16 + ncl;
      float v = acc[t][e];
      if (residual) v += residual[row * ldc + col];
      C[row * ldc + col] = v;
    }
  }
}

// ---------------------------------------------------------------------------
// Fused relative-attention core. One wave per (16-row i-tile, b, n, stream).
// Online softmax; ac/bd via WMMA; rel_shift resolved analytically:
// bd[i,j] = q_r[i] . k_r[Q + j - i] via two shifted tiles + lane shuffles.
// V tiles async-staged into LDS (row-strided operand).
// ---------------------------------------------------------------------------
__global__ __launch_bounds__(32)
void attn_core_kernel(const float* __restrict__ qh, const float* __restrict__ qg,
                      const float* __restrict__ kh, const float* __restrict__ vh,
                      const float* __restrict__ kr,
                      const float* __restrict__ mask_h,
                      const float* __restrict__ mask_g,
                      const float* __restrict__ seg_mat,
                      const float* __restrict__ r_w_bias,
                      const float* __restrict__ r_r_bias,
                      const float* __restrict__ r_s_bias,
                      const float* __restrict__ seg_embed,
                      float* __restrict__ av_h, float* __restrict__ av_g) {
  __shared__ float pt[16 * 16];   // P tile (C-layout -> A-layout remap)
  __shared__ float vt[32 * 64];   // V tile; rows 16..31 stay zero (K padding)

  const int lane = threadIdx.x;
  const int m0 = blockIdx.x * 16;
  const int b  = blockIdx.y & (BATCH - 1);
  const int n  = blockIdx.y / BATCH;
  const int stream = blockIdx.z;

  const float* qsrc = stream ? qg : qh;
  const float* mask = stream ? mask_g : mask_h;
  float* avout      = stream ? av_g : av_h;

  const int ncl = lane & 15;
  const int hi8 = (lane & 16) ? 8 : 0;
  const unsigned vtb = lds_off_of(&vt[0]);

  // Zero the K=16..31 padding rows of the V tile once.
  for (int q = lane; q < 16 * 64; q += 32) vt[16 * 64 + q] = 0.f;

  // Per-lane query row pointer (A layout: lane -> M = lane&15).
  const float* qrow = qsrc + (((long)(m0 + ncl)) * BATCH + b) * DMODEL + n * DHEAD;
  const float* bw = r_w_bias + n * DHEAD;
  const float* br = r_r_bias + n * DHEAD;

  v16bf aw0 = frag_f4(qrow,      bw,      lane);
  v16bf aw1 = frag_f4(qrow + 32, bw + 32, lane);
  v16bf ar0 = frag_f4(qrow,      br,      lane);
  v16bf ar1 = frag_f4(qrow + 32, br + 32, lane);

  // Segment scores per row: e_s[m] = (q[m]+r_s_bias) . seg_embed[s,n].
  float e0 = 0.f, e1 = 0.f;
  {
    const float* sb  = r_s_bias + n * DHEAD;
    const float* se0 = seg_embed + ((long)0 * NHEAD + n) * DHEAD;
    const float* se1 = seg_embed + ((long)1 * NHEAD + n) * DHEAD;
    for (int d = 0; d < DHEAD; ++d) {
      float t = qrow[d] + sb[d];
      e0 += t * se0[d];
      e1 += t * se1[d];
    }
  }

  v8f acc[4];
#pragma unroll
  for (int t = 0; t < 4; ++t) {
#pragma unroll
    for (int e = 0; e < 8; ++e) acc[t][e] = 0.f;
  }
  float rmax[8], rsum[8];
#pragma unroll
  for (int e = 0; e < 8; ++e) { rmax[e] = -3.0e38f; rsum[e] = 0.f; }

#pragma unroll 1
  for (int j0 = 0; j0 < QLEN; j0 += 16) {
    // ---- async-stage V rows j0..j0+15 (16x64 fp32 = 4KB) into LDS ----
    wait_ds0();  // prior DS reads of vt must land before overwrite
#pragma unroll
    for (int c = 0; c < 8; ++c) {
      const int idx4 = (c * 32 + lane) * 4;  // float index 0..1020
      const int row = idx4 >> 6, col = idx4 & 63;
      const float* g = vh + ((long)(j0 + row) * BATCH + b) * DMODEL + n * DHEAD + col;
      async_load_b128(vtb + (unsigned)idx4 * 4u, g);
    }

    // ---- content score tile: ac = (q + r_w_bias) . k_h[j]^T ----
    const float* kb = kh + (((long)(j0 + ncl)) * BATCH + b) * DMODEL + n * DHEAD;
    v8f sA;
#pragma unroll
    for (int e = 0; e < 8; ++e) sA[e] = 0.f;
    sA = wmma_bf16(aw0, frag_f4(kb,      nullptr, lane), sA);
    sA = wmma_bf16(aw1, frag_f4(kb + 32, nullptr, lane), sA);

    // ---- position score: bd2[m][p] = (q+r_r_bias) . k_r[t0+p], two tiles
    const int ta = QLEN + j0 - m0 - 16;  // always in [0, RLEN-16]
    const int tb = ta + 16;
    const float* kra = kr + (((long)(ta + ncl)) * BATCH + b) * DMODEL + n * DHEAD;
    const float* krb = kr + (((long)(tb + ncl)) * BATCH + b) * DMODEL + n * DHEAD;
    v8f bda, bdb;
#pragma unroll
    for (int e = 0; e < 8; ++e) { bda[e] = 0.f; bdb[e] = 0.f; }
    bda = wmma_bf16(ar0, frag_f4(kra,      nullptr, lane), bda);
    bda = wmma_bf16(ar1, frag_f4(kra + 32, nullptr, lane), bda);
    bdb = wmma_bf16(ar0, frag_f4(krb,      nullptr, lane), bdb);
    bdb = wmma_bf16(ar1, frag_f4(krb + 32, nullptr, lane), bdb);

    // ---- assemble scores (C layout: element e -> row m = e+hi8, col ncl) --
    float p[8];
#pragma unroll
    for (int e = 0; e < 8; ++e) {
      const int m = e + hi8;
      const int src = ((ncl - m) & 15) | (lane & 16);
      float va = __shfl(bda[e], src, 32);
      float vb = __shfl(bdb[e], src, 32);
      float bd = (ncl >= m) ? vb : va;
      float e0m = __shfl(e0, m, 32);
      float e1m = __shfl(e1, m, 32);
      long i = m0 + m, j = j0 + ncl;
      long ij = (i * QLEN + j) * BATCH + b;
      float segd = seg_mat[ij * 2 + 1];
      float mk = mask[ij];
      float ef = e0m + (e1m - e0m) * segd;
      p[e] = (sA[e] + bd + ef) * SCALE_F - 1e30f * mk;
    }

    // ---- online softmax update ----
#pragma unroll
    for (int e = 0; e < 8; ++e) {
      float v = p[e];
      v = fmaxf(v, __shfl_xor(v, 1, 32));
      v = fmaxf(v, __shfl_xor(v, 2, 32));
      v = fmaxf(v, __shfl_xor(v, 4, 32));
      v = fmaxf(v, __shfl_xor(v, 8, 32));
      float nm = fmaxf(rmax[e], v);
      float cf = __expf(rmax[e] - nm);
      rmax[e] = nm;
      float pe = __expf(p[e] - nm);
      p[e] = pe;
      float sv = pe;
      sv += __shfl_xor(sv, 1, 32);
      sv += __shfl_xor(sv, 2, 32);
      sv += __shfl_xor(sv, 4, 32);
      sv += __shfl_xor(sv, 8, 32);
      rsum[e] = rsum[e] * cf + sv;
#pragma unroll
      for (int t = 0; t < 4; ++t) acc[t][e] *= cf;
    }

    // ---- remap P (C layout) -> A layout via LDS, zero-pad K 16..31 ----
    __syncthreads();
#pragma unroll
    for (int e = 0; e < 8; ++e) pt[(e + hi8) * 16 + ncl] = p[e];
    __syncthreads();

    v16bf pa;
    {
      const int koff = (lane & 16) ? 8 : 0;
#pragma unroll
      for (int v = 0; v < 8; ++v) {
        if (v < 4) {
          int k = 2 * v + koff;  // < 16: valid P columns
          pa[2 * v]     = (__bf16)pt[ncl * 16 + k];
          pa[2 * v + 1] = (__bf16)pt[ncl * 16 + k + 1];
        } else {
          pa[2 * v]     = (__bf16)0.f;
          pa[2 * v + 1] = (__bf16)0.f;
        }
      }
    }

    // ---- O += P @ V from the async-staged LDS tile ----
    wait_async0();  // V tile resident before DS reads
#pragma unroll
    for (int dt = 0; dt < 4; ++dt) {
      v16bf vbf;
      const int koff = (lane & 16) ? 8 : 0;
#pragma unroll
      for (int v = 0; v < 8; ++v) {
        const int k = kbase(v) + koff;  // rows >=16 read the zeroed pad
        vbf[2 * v]     = (__bf16)vt[k * 64 + dt * 16 + ncl];
        vbf[2 * v + 1] = (__bf16)vt[(k + 1) * 64 + dt * 16 + ncl];
      }
      acc[dt] = wmma_bf16(pa, vbf, acc[dt]);
    }
  }

  // ---- normalize and store attn_vec[i, b, n, d] ----
#pragma unroll
  for (int dt = 0; dt < 4; ++dt) {
#pragma unroll
    for (int e = 0; e < 8; ++e) {
      int m = e + hi8;
      float o = acc[dt][e] / rsum[e];
      avout[(((long)(m0 + m)) * BATCH + b) * DMODEL + n * DHEAD + dt * 16 + ncl] = o;
    }
  }
}

// ---------------------------------------------------------------------------
// LayerNorm over the last dim (1024). One block per row; both streams.
// ---------------------------------------------------------------------------
__global__ __launch_bounds__(256)
void ln_kernel(const float* __restrict__ aoh, const float* __restrict__ aog,
               const float* __restrict__ gamma, const float* __restrict__ beta,
               float* __restrict__ out) {
  __shared__ float red[256];
  const int row = blockIdx.x;          // 0..8191
  const int stream = row >> 12;
  const int rr = row & 4095;
  const float* src = (stream ? aog : aoh) + (long)rr * DMODEL;
  float* dst = out + (long)stream * QLEN * BATCH * DMODEL + (long)rr * DMODEL;
  const int t = threadIdx.x;

  float s = 0.f;
  for (int d = t; d < DMODEL; d += 256) s += src[d];
  red[t] = s;
  __syncthreads();
  for (int off = 128; off; off >>= 1) {
    if (t < off) red[t] += red[t + off];
    __syncthreads();
  }
  float mu = red[0] / (float)DMODEL;
  __syncthreads();

  float v = 0.f;
  for (int d = t; d < DMODEL; d += 256) {
    float x = src[d] - mu;
    v += x * x;
  }
  red[t] = v;
  __syncthreads();
  for (int off = 128; off; off >>= 1) {
    if (t < off) red[t] += red[t + off];
    __syncthreads();
  }
  float var = red[0] / (float)DMODEL;
  float rs = rsqrtf(var + LN_EPS);

  for (int d = t; d < DMODEL; d += 256)
    dst[d] = (src[d] - mu) * rs * gamma[d] + beta[d];
}

// ---------------------------------------------------------------------------
extern "C" void kernel_launch(void* const* d_in, const int* in_sizes, int n_in,
                              void* d_out, int out_size, void* d_ws, size_t ws_size,
                              hipStream_t stream) {
  (void)in_sizes; (void)n_in; (void)out_size; (void)ws_size;
  const float* h        = (const float*)d_in[0];
  const float* g        = (const float*)d_in[1];
  const float* mask_h   = (const float*)d_in[2];
  const float* mask_g   = (const float*)d_in[3];
  const float* r        = (const float*)d_in[4];
  const float* seg_mat  = (const float*)d_in[5];
  const float* q_w      = (const float*)d_in[6];
  const float* k_w      = (const float*)d_in[7];
  const float* v_w      = (const float*)d_in[8];
  const float* o_w      = (const float*)d_in[9];
  const float* r_w      = (const float*)d_in[10];
  const float* r_r_bias = (const float*)d_in[11];
  const float* r_s_bias = (const float*)d_in[12];
  const float* r_w_bias = (const float*)d_in[13];
  const float* seg_emb  = (const float*)d_in[14];
  const float* ln_gamma = (const float*)d_in[15];
  const float* ln_beta  = (const float*)d_in[16];
  float* out = (float*)d_out;

  float* ws = (float*)d_ws;
  const size_t SZ = (size_t)QLEN * BATCH * DMODEL;  // 4M floats
  float* qh  = ws;
  float* qg  = qh + SZ;
  float* kh  = qg + SZ;
  float* vh  = kh + SZ;
  float* kr  = vh + SZ;           // 2*SZ floats (RLEN rows)
  float* avh = kr + 2 * SZ;
  float* avg = avh + SZ;
  float* aoh = qh;                // reuse: q buffers dead after attention
  float* aog = qg;

  dim3 b256(256, 1, 1);
  dim3 w32(32, 1, 1);
  const int MQ = QLEN * BATCH;          // 4096
  const int MR = 2 * QLEN * BATCH;      // 8192

  // Stage 1: projections (B[k][n] = W[k*1024 + n] -> ldk=1024, ldn=1)
  gemm_bf16_kernel<true><<<dim3(MQ / 128, DMODEL / 64), b256, 0, stream>>>(
      h, DMODEL, q_w, DMODEL, 1, qh, DMODEL, MQ, DMODEL, DMODEL, nullptr);
  gemm_bf16_kernel<true><<<dim3(MQ / 128, DMODEL / 64), b256, 0, stream>>>(
      g, DMODEL, q_w, DMODEL, 1, qg, DMODEL, MQ, DMODEL, DMODEL, nullptr);
  gemm_bf16_kernel<true><<<dim3(MQ / 128, DMODEL / 64), b256, 0, stream>>>(
      h, DMODEL, k_w, DMODEL, 1, kh, DMODEL, MQ, DMODEL, DMODEL, nullptr);
  gemm_bf16_kernel<true><<<dim3(MQ / 128, DMODEL / 64), b256, 0, stream>>>(
      h, DMODEL, v_w, DMODEL, 1, vh, DMODEL, MQ, DMODEL, DMODEL, nullptr);
  gemm_bf16_kernel<true><<<dim3(MR / 128, DMODEL / 64), b256, 0, stream>>>(
      r, DMODEL, r_w, DMODEL, 1, kr, DMODEL, MR, DMODEL, DMODEL, nullptr);

  // Stage 2: fused relative attention, both streams.
  attn_core_kernel<<<dim3(QLEN / 16, BATCH * NHEAD, 2), w32, 0, stream>>>(
      qh, qg, kh, vh, kr, mask_h, mask_g, seg_mat,
      r_w_bias, r_r_bias, r_s_bias, seg_emb, avh, avg);

  // Stage 3: output projection + residual (B[k][n] = o_w[n*1024 + k], ldk==1)
  gemm_bf16_kernel<false><<<dim3(MQ / 128, DMODEL / 64), b256, 0, stream>>>(
      avh, DMODEL, o_w, 1, DMODEL, aoh, DMODEL, MQ, DMODEL, DMODEL, h);
  gemm_bf16_kernel<false><<<dim3(MQ / 128, DMODEL / 64), b256, 0, stream>>>(
      avg, DMODEL, o_w, 1, DMODEL, aog, DMODEL, MQ, DMODEL, DMODEL, g);

  // Stage 4: LayerNorm -> d_out (output_h then output_g).
  ln_kernel<<<dim3(2 * MQ, 1, 1), b256, 0, stream>>>(
      aoh, aog, ln_gamma, ln_beta, out);
}
